// STTLayer_80470507258275
// MI455X (gfx1250) — compile-verified, matrix operand order checked by reading, not run on hardware
//
#include <hip/hip_runtime.h>
#include <hip/hip_bf16.h>

// ---------------------------------------------------------------------------
// STT layer for gfx1250 (MI455X): bf16 WMMA everywhere matmuls matter.
// Weights are transposed at conversion so all WMMA fragments load as
// contiguous b128s; V is stored transposed per head for the same reason.
// Attention stages K/V^T tiles through LDS with async loads (double buffer).
// ---------------------------------------------------------------------------

typedef __bf16 bf16_t;
typedef __attribute__((ext_vector_type(16))) __bf16 v16bf;
typedef __attribute__((ext_vector_type(8)))  __bf16 v8bf;
typedef __attribute__((ext_vector_type(8)))  float  v8f;

#define B_    4
#define T_    4096
#define D_    2048
#define H_    16
#define HD_   128
#define DFF_  5632
#define KSEL_ 2048
#define M_    (B_ * KSEL_)     // 8192 selected rows total
#define EPS_  1e-6f
#define CH_   32               // attention kv chunk

// Load a WMMA 16-bit fragment: two contiguous 16B chunks (K-halves per lane).
__device__ inline v16bf ldfrag(const bf16_t* __restrict__ p) {
  v8bf lo = *(const v8bf*)p;
  v8bf hi = *(const v8bf*)(p + 16);
  return __builtin_shufflevector(lo, hi, 0, 1, 2, 3, 4, 5, 6, 7, 8, 9, 10, 11,
                                 12, 13, 14, 15);
}

// Async global -> LDS copy of 16 bytes (per lane). ASYNCcnt-tracked.
__device__ inline void async_b128(unsigned lds_off, const bf16_t* g) {
  asm volatile("global_load_async_to_lds_b128 %0, %1, off"
               :: "v"(lds_off), "v"(g) : "memory");
}

// ======================= router: logits = x . w_router =====================
__global__ __launch_bounds__(256) void router_kernel(
    const float* __restrict__ hidden, const float* __restrict__ wr,
    float* __restrict__ logits) {
  int wid  = (blockIdx.x * 256 + threadIdx.x) >> 5;   // one wave per token
  int lane = threadIdx.x & 31;
  if (wid >= B_ * T_) return;
  const float* x = hidden + (long)wid * D_;
  float s = 0.f;
  for (int i = lane; i < D_; i += 32) s += x[i] * wr[i];
  #pragma unroll
  for (int d = 1; d < 32; d <<= 1) s += __shfl_xor(s, d, 32);
  if (lane == 0) logits[wid] = s;
}

// ============== top-k by rank (matches jax.lax.top_k tie-break) ============
__global__ __launch_bounds__(256) void rank_kernel(
    const float* __restrict__ logits, unsigned char* __restrict__ flags) {
  int i = blockIdx.x * 256 + threadIdx.x;
  int b = blockIdx.y;
  if (i >= T_) return;
  const float* lg = logits + (long)b * T_;
  float li = lg[i];
  int rank = 0;
  for (int j = 0; j < T_; ++j) {
    float lj = lg[j];
    rank += (lj > li) || (lj == li && j < i);
  }
  flags[(long)b * T_ + i] = (rank < KSEL_) ? 1 : 0;
}

__global__ void compact_kernel(const unsigned char* __restrict__ flags,
                               int* __restrict__ idx) {
  int b = threadIdx.x;
  if (b >= B_) return;
  int pos = 0;
  for (int j = 0; j < T_; ++j)
    if (flags[(long)b * T_ + j]) idx[(long)b * KSEL_ + (pos++)] = j;
}

// ============== fp32 [K,N] -> bf16 transposed [N,K] ========================
__global__ __launch_bounds__(256) void f2bf_t_kernel(
    const float* __restrict__ in, bf16_t* __restrict__ out, int K, int N) {
  long n = (long)K * N;
  long i = (long)blockIdx.x * 256 + threadIdx.x;
  long stride = (long)gridDim.x * 256;
  for (; i < n; i += stride) {
    long nn = i / K, kk = i % K;
    out[i] = (bf16_t)in[kk * N + nn];
  }
}

// ============ RMSNorm (gather variant from hidden via idx) -> bf16 =========
__global__ __launch_bounds__(256) void rms_gather_kernel(
    const float* __restrict__ hidden, const int* __restrict__ idx,
    const float* __restrict__ w, bf16_t* __restrict__ out) {
  int s = blockIdx.x, b = blockIdx.y;
  int t = idx[(long)b * KSEL_ + s];
  const float* x = hidden + ((long)b * T_ + t) * D_;
  __shared__ float red[8];
  float ss = 0.f;
  for (int i = threadIdx.x; i < D_; i += 256) { float v = x[i]; ss += v * v; }
  #pragma unroll
  for (int d = 1; d < 32; d <<= 1) ss += __shfl_xor(ss, d, 32);
  if ((threadIdx.x & 31) == 0) red[threadIdx.x >> 5] = ss;
  __syncthreads();
  float tot = red[0] + red[1] + red[2] + red[3] + red[4] + red[5] + red[6] + red[7];
  float inv = rsqrtf(tot / (float)D_ + EPS_);
  bf16_t* o = out + ((long)b * KSEL_ + s) * D_;
  for (int i = threadIdx.x; i < D_; i += 256) o[i] = (bf16_t)(x[i] * inv * w[i]);
}

// ============ RMSNorm (plain fp32 rows) -> bf16 ============================
__global__ __launch_bounds__(256) void rms_plain_kernel(
    const float* __restrict__ X, const float* __restrict__ w,
    bf16_t* __restrict__ out) {
  long row = blockIdx.x;
  const float* x = X + row * D_;
  __shared__ float red[8];
  float ss = 0.f;
  for (int i = threadIdx.x; i < D_; i += 256) { float v = x[i]; ss += v * v; }
  #pragma unroll
  for (int d = 1; d < 32; d <<= 1) ss += __shfl_xor(ss, d, 32);
  if ((threadIdx.x & 31) == 0) red[threadIdx.x >> 5] = ss;
  __syncthreads();
  float tot = red[0] + red[1] + red[2] + red[3] + red[4] + red[5] + red[6] + red[7];
  float inv = rsqrtf(tot / (float)D_ + EPS_);
  bf16_t* o = out + row * D_;
  for (int i = threadIdx.x; i < D_; i += 256) o[i] = (bf16_t)(x[i] * inv * w[i]);
}

// ======================= WMMA bf16 GEMM ====================================
// C[M,N] = A[M,K] (row-major bf16) x B (given TRANSPOSED: BT[N,K] row-major).
// Block = 4 waves; block tile 32(M) x 256(N); wave tile 32 x 64 (8 WMMA accs).
// mode 0: outF = acc (+ resid)                (fp32 out)
// mode 1: outB = bf16(acc)
// mode 2: outB = bf16(silu(acc))
// mode 3: outB = bf16(acc * mul[m,n])         (mul bf16, ld = N)
// mode 4: V-transpose store: vt[b][col][s] = bf16(acc), KSEL/D hardcoded
__global__ __launch_bounds__(128) void gemm_wmma_bf16(
    const bf16_t* __restrict__ A, const bf16_t* __restrict__ BT,
    int M, int N, int K,
    float* __restrict__ outF, bf16_t* __restrict__ outB,
    const float* __restrict__ resid, const bf16_t* __restrict__ mul,
    int mode) {
  int lane = threadIdx.x & 31;
  int wave = threadIdx.x >> 5;
  int m0 = blockIdx.x * 32;
  int n0 = blockIdx.y * 256 + wave * 64;
  int klo = (lane >> 4) << 3;                 // 0 or 8 (fragment half split)
  const bf16_t* pa0 = A + (long)(m0 + (lane & 15)) * K + klo;
  const bf16_t* pa1 = pa0 + (long)16 * K;
  const bf16_t* pb = BT + (long)(n0 + (lane & 15)) * K + klo;
  v8f acc[2][4];
  #pragma unroll
  for (int mi = 0; mi < 2; ++mi)
    #pragma unroll
    for (int j = 0; j < 4; ++j) acc[mi][j] = (v8f){};

  for (int k0 = 0; k0 < K; k0 += 32) {
    v16bf a0 = ldfrag(pa0);
    v16bf a1 = ldfrag(pa1);
    __builtin_prefetch(pa0 + 64, 0, 0);
    #pragma unroll
    for (int j = 0; j < 4; ++j) {
      v16bf b = ldfrag(pb + (long)j * 16 * K);
      acc[0][j] = __builtin_amdgcn_wmma_f32_16x16x32_bf16(
          false, a0, false, b, (short)0, acc[0][j], false, false);
      acc[1][j] = __builtin_amdgcn_wmma_f32_16x16x32_bf16(
          false, a1, false, b, (short)0, acc[1][j], false, false);
    }
    pa0 += 32; pa1 += 32; pb += 32;
  }

  #pragma unroll
  for (int mi = 0; mi < 2; ++mi) {
    int rbase = m0 + mi * 16 + ((lane >> 4) << 3);
    #pragma unroll
    for (int j = 0; j < 4; ++j) {
      int col = n0 + j * 16 + (lane & 15);
      #pragma unroll
      for (int r = 0; r < 8; ++r) {
        int row = rbase + r;
        long off = (long)row * N + col;
        float v = acc[mi][j][r];
        if (mode == 0) {
          if (resid) v += resid[off];
          outF[off] = v;
        } else if (mode == 1) {
          outB[off] = (bf16_t)v;
        } else if (mode == 2) {
          outB[off] = (bf16_t)(v / (1.f + __expf(-v)));   // silu
        } else if (mode == 3) {
          outB[off] = (bf16_t)(v * (float)mul[off]);
        } else {                                          // mode 4: V^T store
          int b = row >> 11;                              // row / KSEL_
          int s = row & (KSEL_ - 1);
          outB[((long)b * D_ + col) * KSEL_ + s] = (bf16_t)v;
        }
      }
    }
  }
}

// ======================= RoPE (original positions) =========================
__global__ __launch_bounds__(256) void rope_kernel(
    const float* __restrict__ qf, const float* __restrict__ kf,
    const int* __restrict__ idx, bf16_t* __restrict__ qb,
    bf16_t* __restrict__ kb) {
  long gid = (long)blockIdx.x * 256 + threadIdx.x;  // over B*KSEL*H*64
  if (gid >= (long)M_ * H_ * 64) return;
  int i = gid & 63;
  long tmp = gid >> 6;
  int h = (int)(tmp % H_); tmp /= H_;
  int s = (int)(tmp % KSEL_);
  int b = (int)(tmp / KSEL_);
  float pos = (float)idx[(long)b * KSEL_ + s];
  float freq = __powf(10000.0f, -2.0f * (float)i / (float)HD_);
  float ang = pos * freq;
  float c = __cosf(ang), sn = __sinf(ang);
  long base = ((long)b * KSEL_ + s) * D_ + (long)h * HD_;
  float q1 = qf[base + i], q2 = qf[base + i + 64];
  qb[base + i]      = (bf16_t)(q1 * c - q2 * sn);
  qb[base + i + 64] = (bf16_t)(q2 * c + q1 * sn);
  float k1 = kf[base + i], k2 = kf[base + i + 64];
  kb[base + i]      = (bf16_t)(k1 * c - k2 * sn);
  kb[base + i + 64] = (bf16_t)(k2 * c + k1 * sn);
}

// ====== Flash attention: wave per (b, h, 16-row q tile), WMMA bf16 =========
// Q,K: [B,KSEL,D] bf16 (head-major inside D). Vt: [B][D][KSEL] bf16.
// K and V^T tiles are staged into LDS with async loads, double-buffered.
__global__ __launch_bounds__(32) void attn_wmma(
    const bf16_t* __restrict__ Q, const bf16_t* __restrict__ Kb,
    const bf16_t* __restrict__ Vt, bf16_t* __restrict__ O) {
  __shared__ bf16_t ldsK[2][CH_][HD_];     // 2 x 8KB
  __shared__ bf16_t ldsV[2][HD_][CH_];     // 2 x 8KB (V^T layout: hd x kv)
  __shared__ bf16_t ldsP[16 * CH_];        // 1KB
  int lane = threadIdx.x;
  int qt = blockIdx.x, h = blockIdx.y, b = blockIdx.z;
  int klo = (lane >> 4) << 3;
  int qrow = qt * 16 + (lane & 15);
  long headBase = (long)b * KSEL_ * D_ + (long)h * HD_;
  long vtBase = ((long)b * D_ + (long)h * HD_) * KSEL_;

  // async-stage one kv chunk (K tile 32x128, V^T tile 128x32) into buffer bb
  auto stage = [&](int bb, int kv0) {
    const bf16_t* gk = Kb + headBase + (long)kv0 * D_;
    unsigned lk = (unsigned)(size_t)&ldsK[bb][0][0];
    #pragma unroll
    for (int i = 0; i < 16; ++i) {           // 512 x 16B, 16 per lane
      int c = i * 32 + lane;
      int row = c >> 4, col = (c & 15) << 3;
      async_b128(lk + (unsigned)(row * HD_ + col) * 2,
                 gk + (long)row * D_ + col);
    }
    const bf16_t* gv = Vt + vtBase + kv0;
    unsigned lv = (unsigned)(size_t)&ldsV[bb][0][0];
    #pragma unroll
    for (int i = 0; i < 16; ++i) {           // 512 x 16B, 16 per lane
      int c = i * 32 + lane;
      int row = c >> 2, col = (c & 3) << 3;
      async_b128(lv + (unsigned)(row * CH_ + col) * 2,
                 gv + (long)row * KSEL_ + col);
    }
  };

  v16bf qa[4];
  #pragma unroll
  for (int c = 0; c < 4; ++c)
    qa[c] = ldfrag(Q + headBase + (long)qrow * D_ + c * 32 + klo);

  float mrow[8], lrow[8];
  v8f acc[8];
  #pragma unroll
  for (int r = 0; r < 8; ++r) { mrow[r] = -1e30f; lrow[r] = 0.f; }
  #pragma unroll
  for (int j = 0; j < 8; ++j) acc[j] = (v8f){};

  const float scale = 0.08838834764831845f;   // 1/sqrt(128)
  int qhi = qt * 16 + 15;
  int rbase = qt * 16 + ((lane >> 4) << 3);
  int nchunks = qhi / CH_ + 1;

  stage(0, 0);
  for (int ci = 0; ci < nchunks; ++ci) {
    int kv0 = ci * CH_;
    int bb = ci & 1;
    asm volatile("s_wait_asynccnt 0x0" ::: "memory");   // buffer bb resident
    if (ci + 1 < nchunks) stage(bb ^ 1, kv0 + CH_);     // overlap next copy

    // ---- S = Q K^T over two 16-col tiles (K^T B-frag from LDS rows)
    v8f s0 = (v8f){}, s1 = (v8f){};
    #pragma unroll
    for (int c = 0; c < 4; ++c) {
      v16bf kb0 = ldfrag(&ldsK[bb][lane & 15][c * 32 + klo]);
      v16bf kb1 = ldfrag(&ldsK[bb][16 + (lane & 15)][c * 32 + klo]);
      s0 = __builtin_amdgcn_wmma_f32_16x16x32_bf16(
          false, qa[c], false, kb0, (short)0, s0, false, false);
      s1 = __builtin_amdgcn_wmma_f32_16x16x32_bf16(
          false, qa[c], false, kb1, (short)0, s1, false, false);
    }
    // ---- online softmax over this 32-col chunk
    int col0 = kv0 + (lane & 15);
    int col1 = col0 + 16;
    float p0v[8], p1v[8];
    #pragma unroll
    for (int r = 0; r < 8; ++r) {
      int row = rbase + r;
      float a0 = (col0 <= row) ? s0[r] * scale : -1e30f;
      float a1 = (col1 <= row) ? s1[r] * scale : -1e30f;
      float mx = fmaxf(a0, a1);
      #pragma unroll
      for (int d = 1; d < 16; d <<= 1) mx = fmaxf(mx, __shfl_xor(mx, d, 32));
      float mnew = fmaxf(mrow[r], mx);
      float f = __expf(mrow[r] - mnew);
      mrow[r] = mnew;
      float e0 = __expf(a0 - mnew);
      float e1 = __expf(a1 - mnew);
      float ls = e0 + e1;
      #pragma unroll
      for (int d = 1; d < 16; d <<= 1) ls += __shfl_xor(ls, d, 32);
      lrow[r] = lrow[r] * f + ls;
      #pragma unroll
      for (int j = 0; j < 8; ++j) acc[j][r] *= f;
      p0v[r] = e0; p1v[r] = e1;
    }
    // ---- C-layout P -> LDS (row-major 16x32) -> A-layout fragment
    #pragma unroll
    for (int r = 0; r < 8; ++r) {
      int rr = r + ((lane >> 4) << 3);
      ldsP[rr * CH_ + (lane & 15)]      = (bf16_t)p0v[r];
      ldsP[rr * CH_ + 16 + (lane & 15)] = (bf16_t)p1v[r];
    }
    asm volatile("s_wait_dscnt 0" ::: "memory");
    v16bf pa = ldfrag(ldsP + (lane & 15) * CH_ + klo);
    // ---- O += P V  (V^T rows are kv-contiguous in LDS)
    #pragma unroll
    for (int j = 0; j < 8; ++j) {
      v16bf vb = ldfrag(&ldsV[bb][j * 16 + (lane & 15)][klo]);
      acc[j] = __builtin_amdgcn_wmma_f32_16x16x32_bf16(
          false, pa, false, vb, (short)0, acc[j], false, false);
    }
    asm volatile("s_wait_dscnt 0" ::: "memory");  // LDS reads done before overwrite
  }
  // ---- epilogue: O = acc / l  (bf16 for the O-projection GEMM)
  #pragma unroll
  for (int j = 0; j < 8; ++j) {
    #pragma unroll
    for (int r = 0; r < 8; ++r) {
      int row = rbase + r;
      float ov = acc[j][r] / lrow[r];
      O[headBase + (long)row * D_ + j * 16 + (lane & 15)] = (bf16_t)ov;
    }
  }
}

// ================= x2 += gather(hidden, idx) (residual add) ================
__global__ __launch_bounds__(256) void addgather_kernel(
    float* __restrict__ x2, const float* __restrict__ hidden,
    const int* __restrict__ idx) {
  long gid = (long)blockIdx.x * 256 + threadIdx.x;
  if (gid >= (long)M_ * D_) return;
  int d = (int)(gid % D_);
  long rs = gid / D_;
  int s = (int)(rs % KSEL_);
  int b = (int)(rs / KSEL_);
  int t = idx[(long)b * KSEL_ + s];
  x2[gid] += hidden[((long)b * T_ + t) * D_ + d];
}

// ======================= out = hidden (copy) ===============================
__global__ __launch_bounds__(256) void copy_kernel(
    const float* __restrict__ in, float* __restrict__ out, long n) {
  long i = (long)blockIdx.x * 256 + threadIdx.x;
  if (i < n) out[i] = in[i];
}

// ======================= scatter processed rows ============================
__global__ __launch_bounds__(256) void scatter_kernel(
    float* __restrict__ out, const float* __restrict__ x2,
    const int* __restrict__ idx) {
  long gid = (long)blockIdx.x * 256 + threadIdx.x;
  if (gid >= (long)M_ * D_) return;
  int d = (int)(gid % D_);
  long rs = gid / D_;
  int s = (int)(rs % KSEL_);
  int b = (int)(rs / KSEL_);
  int t = idx[(long)b * KSEL_ + s];
  out[((long)b * T_ + t) * D_ + d] = x2[gid];
}

// ===========================================================================
extern "C" void kernel_launch(void* const* d_in, const int* in_sizes, int n_in,
                              void* d_out, int out_size, void* d_ws,
                              size_t ws_size, hipStream_t stream) {
  const float* hidden = (const float*)d_in[0];
  const float* w_router = (const float*)d_in[1];
  const float* rms1_w = (const float*)d_in[2];
  const float* wq = (const float*)d_in[3];
  const float* wk = (const float*)d_in[4];
  const float* wv = (const float*)d_in[5];
  const float* wo = (const float*)d_in[6];
  const float* rms2_w = (const float*)d_in[7];
  const float* w_gate = (const float*)d_in[8];
  const float* w_up = (const float*)d_in[9];
  const float* w_down = (const float*)d_in[10];
  float* out = (float*)d_out;
  char* ws = (char*)d_ws;

  // ---- workspace arena (with reuse) ----
  const size_t o_logits = 0;                                  // B*T fp32
  const size_t o_flags = o_logits + (size_t)B_ * T_ * 4;      // B*T u8
  const size_t o_idx = o_flags + (size_t)B_ * T_;             // B*KSEL i32
  size_t cur = 1ull << 20;
  const size_t o_wq = cur; cur += (size_t)D_ * D_ * 2;        // transposed bf16
  const size_t o_wk = cur; cur += (size_t)D_ * D_ * 2;
  const size_t o_wv = cur; cur += (size_t)D_ * D_ * 2;
  const size_t o_wo = cur; cur += (size_t)D_ * D_ * 2;
  const size_t o_wg = cur; cur += (size_t)D_ * DFF_ * 2;
  const size_t o_wu = cur; cur += (size_t)D_ * DFF_ * 2;
  const size_t o_wd = cur; cur += (size_t)DFF_ * D_ * 2;
  const size_t o_R1 = cur; cur += (size_t)M_ * D_ * 4;  // qf  -> o_bf | h2_bf
  const size_t o_R2 = cur; cur += (size_t)M_ * D_ * 4;  // kf  -> x2
  const size_t o_R3 = cur; cur += (size_t)M_ * D_ * 2;  // h1  -> qb
  const size_t o_R4 = cur; cur += (size_t)M_ * D_ * 2;  // kb
  const size_t o_R5 = cur; cur += (size_t)M_ * D_ * 2;  // vt
  const size_t o_R6 = cur; cur += (size_t)M_ * DFF_ * 2;  // gate-silu bf16
  const size_t o_R7 = cur; cur += (size_t)M_ * DFF_ * 2;  // act bf16
  (void)ws_size; (void)n_in; (void)in_sizes; (void)out_size;

  float* logits = (float*)(ws + o_logits);
  unsigned char* flags = (unsigned char*)(ws + o_flags);
  int* idx = (int*)(ws + o_idx);
  bf16_t* wq_t = (bf16_t*)(ws + o_wq);
  bf16_t* wk_t = (bf16_t*)(ws + o_wk);
  bf16_t* wv_t = (bf16_t*)(ws + o_wv);
  bf16_t* wo_t = (bf16_t*)(ws + o_wo);
  bf16_t* wg_t = (bf16_t*)(ws + o_wg);
  bf16_t* wu_t = (bf16_t*)(ws + o_wu);
  bf16_t* wd_t = (bf16_t*)(ws + o_wd);
  float* qf = (float*)(ws + o_R1);
  float* kf = (float*)(ws + o_R2);
  bf16_t* h1 = (bf16_t*)(ws + o_R3);
  bf16_t* qb = (bf16_t*)(ws + o_R3);                    // reuse after h1 dead
  bf16_t* kb = (bf16_t*)(ws + o_R4);
  bf16_t* vt = (bf16_t*)(ws + o_R5);
  bf16_t* ob = (bf16_t*)(ws + o_R1);                    // reuse qf (dead)
  bf16_t* h2 = (bf16_t*)(ws + o_R1 + (size_t)M_ * D_ * 2);
  float* x2 = (float*)(ws + o_R2);                      // reuse kf (dead)
  bf16_t* gs = (bf16_t*)(ws + o_R6);
  bf16_t* ab = (bf16_t*)(ws + o_R7);

  // 1) router logits
  router_kernel<<<dim3(B_ * T_ / 8), dim3(256), 0, stream>>>(hidden, w_router,
                                                             logits);
  // 2) top-k selection by rank, then compact (sorted ascending)
  rank_kernel<<<dim3(T_ / 256, B_), dim3(256), 0, stream>>>(logits, flags);
  compact_kernel<<<dim3(1), dim3(32), 0, stream>>>(flags, idx);

  // 3) weights -> bf16, transposed to [N,K] for contiguous B-fragments
  f2bf_t_kernel<<<dim3(4096), dim3(256), 0, stream>>>(wq, wq_t, D_, D_);
  f2bf_t_kernel<<<dim3(4096), dim3(256), 0, stream>>>(wk, wk_t, D_, D_);
  f2bf_t_kernel<<<dim3(4096), dim3(256), 0, stream>>>(wv, wv_t, D_, D_);
  f2bf_t_kernel<<<dim3(4096), dim3(256), 0, stream>>>(wo, wo_t, D_, D_);
  f2bf_t_kernel<<<dim3(4096), dim3(256), 0, stream>>>(w_gate, wg_t, D_, DFF_);
  f2bf_t_kernel<<<dim3(4096), dim3(256), 0, stream>>>(w_up, wu_t, D_, DFF_);
  f2bf_t_kernel<<<dim3(4096), dim3(256), 0, stream>>>(w_down, wd_t, DFF_, D_);

  // 4) gather + RMSNorm1 -> h1 (bf16)
  rms_gather_kernel<<<dim3(KSEL_, B_), dim3(256), 0, stream>>>(hidden, idx,
                                                               rms1_w, h1);

  // 5) Q/K/V projections (WMMA): q,k fp32 for RoPE; v directly to V^T bf16
  dim3 gD(M_ / 32, D_ / 256);
  gemm_wmma_bf16<<<gD, dim3(128), 0, stream>>>(h1, wq_t, M_, D_, D_, qf,
                                               nullptr, nullptr, nullptr, 0);
  gemm_wmma_bf16<<<gD, dim3(128), 0, stream>>>(h1, wk_t, M_, D_, D_, kf,
                                               nullptr, nullptr, nullptr, 0);
  gemm_wmma_bf16<<<gD, dim3(128), 0, stream>>>(h1, wv_t, M_, D_, D_, nullptr,
                                               vt, nullptr, nullptr, 4);

  // 6) RoPE with original token positions -> bf16 Q/K
  rope_kernel<<<dim3((long)M_ * H_ * 64 / 256), dim3(256), 0, stream>>>(
      qf, kf, idx, qb, kb);

  // 7) flash attention (WMMA + async LDS staging + LDS P round-trip)
  attn_wmma<<<dim3(KSEL_ / 16, H_, B_), dim3(32), 0, stream>>>(qb, kb, vt, ob);

  // 8) O projection -> x2 (fp32), then residual add of gathered tokens
  gemm_wmma_bf16<<<gD, dim3(128), 0, stream>>>(ob, wo_t, M_, D_, D_, x2,
                                               nullptr, nullptr, nullptr, 0);
  addgather_kernel<<<dim3((long)M_ * D_ / 256), dim3(256), 0, stream>>>(
      x2, hidden, idx);

  // 9) RMSNorm2 -> h2 (bf16)
  rms_plain_kernel<<<dim3(M_), dim3(256), 0, stream>>>(x2, rms2_w, h2);

  // 10) MLP: gate (silu epilogue), up (mul epilogue), down (+residual)
  dim3 gF(M_ / 32, DFF_ / 256);
  gemm_wmma_bf16<<<gF, dim3(128), 0, stream>>>(h2, wg_t, M_, DFF_, D_, nullptr,
                                               gs, nullptr, nullptr, 2);
  gemm_wmma_bf16<<<gF, dim3(128), 0, stream>>>(h2, wu_t, M_, DFF_, D_, nullptr,
                                               ab, nullptr, gs, 3);
  gemm_wmma_bf16<<<gD, dim3(128), 0, stream>>>(ab, wd_t, M_, D_, DFF_, x2,
                                               nullptr, x2, nullptr, 0);

  // 11) out = hidden; scatter processed rows back
  copy_kernel<<<dim3((long)B_ * T_ * D_ / 256), dim3(256), 0, stream>>>(
      hidden, out, (long)B_ * T_ * D_);
  scatter_kernel<<<dim3((long)M_ * D_ / 256), dim3(256), 0, stream>>>(out, x2,
                                                                      idx);
}
